// TriangleMultiplication_21225728377411
// MI455X (gfx1250) — matrix operand order repaired and checked by reading, not compiled, not run on hardware
//
#include <hip/hip_runtime.h>
#include <hip/hip_bf16.h>

// ---------------------------------------------------------------------------
// Triangle multiplication (incoming) for MI455X / gfx1250, wave32 + WMMA.
// Heavy einsum 'bkjd,bkid->bijd' = 128 independent 512x512x512 GEMMs, done
// with v_wmma_f32_16x16x32_f16. All other GEMMs also WMMA. The heavy GEMM
// stages K-slabs into LDS with GLOBAL_LOAD_ASYNC_TO_LDS_B64 (ASYNCcnt) when
// the toolchain exposes the builtin, else a sync VGPR round-trip fallback.
// ---------------------------------------------------------------------------

typedef _Float16 h16;
typedef __attribute__((ext_vector_type(2)))  _Float16 h2;
typedef __attribute__((ext_vector_type(4)))  _Float16 h4;
typedef __attribute__((ext_vector_type(8)))  _Float16 v8h;
typedef __attribute__((ext_vector_type(16))) _Float16 v16h;
typedef __attribute__((ext_vector_type(8)))  float    v8f;

#define NSEQ 512
#define DDIM 128
#define MPOS (NSEQ * NSEQ)   // 262144 positions

#if defined(__AMDGCN__) && defined(__has_builtin)
#if __has_builtin(__builtin_amdgcn_global_load_async_to_lds_b64) && \
    __has_builtin(__builtin_amdgcn_s_wait_asynccnt)
#define USE_ASYNC_LDS 1
#endif
#endif
#ifndef USE_ASYNC_LDS
#define USE_ASYNC_LDS 0
#endif

// The async builtin's params are pointers to a gcc-style int2 vector in
// addrspace(1)/(3) (per the clang diagnostic: "vector_size(2*sizeof(int)) int
// __device__ *"). Cast via integer round-trip (LDS generic pointers carry the
// aperture in the high 32 bits; low 32 bits = LDS byte offset).
typedef int async_i2 __attribute__((vector_size(8)));
#define AS1P64(p) ((__attribute__((address_space(1))) async_i2*)(unsigned long long)(p))
#define AS3P64(p) ((__attribute__((address_space(3))) async_i2*)(unsigned int)(unsigned long long)(p))

__device__ __forceinline__ float sigmoidf_(float x) {
    return 1.0f / (1.0f + __expf(-x));
}

__device__ __forceinline__ v8f wmma16(v16h a, v16h b, v8f c) {
    return __builtin_amdgcn_wmma_f32_16x16x32_f16(
        /*neg_a=*/false, a, /*neg_b=*/false, b,
        /*c_mod=*/(short)0, c, /*reuse_a=*/false, /*reuse_b=*/false);
}

// A fragment (16x32, M-row-major source, ld = 128 halves).
// Per ISA 7.12.2: lane<16 holds M=lane, K={k0..k0+7, k0+16..k0+23};
// lane>=16 holds M=lane-16, K={k0+8..k0+15, k0+24..k0+31}.
__device__ __forceinline__ v16h frag_a(const h16* rowp, int k0, int hi) {
    union { v16h v; v8h h[2]; } u;
    u.h[0] = *(const v8h*)(rowp + k0 + hi * 8);
    u.h[1] = *(const v8h*)(rowp + k0 + hi * 8 + 16);
    return u.v;
}

// B fragment (32x16) from an N-major weight buffer: rowp points at row n
// (contiguous K, ld = 128). lane<16 -> K={k0..k0+7, k0+16..k0+23}, lane>=16 -> K={k0+8..k0+15, k0+24..k0+31}.
__device__ __forceinline__ v16h frag_b(const h16* rowp, int k0, int hi) {
    union { v16h v; v8h h[2]; } u;
    u.h[0] = *(const v8h*)(rowp + k0 + hi * 8);
    u.h[1] = *(const v8h*)(rowp + k0 + hi * 8 + 16);
    return u.v;
}

// ---------------------------------------------------------------------------
// k_pack: fp32 weights -> f16, layout [out_channel][K] (N-major, contiguous K)
// rows 0..511 = W_lr, 512..639 = W_gate, 640..767 = W_out
// ---------------------------------------------------------------------------
__global__ void k_pack(const float* __restrict__ wlr,
                       const float* __restrict__ wgate,
                       const float* __restrict__ wout_in,
                       h16* __restrict__ wcat, h16* __restrict__ wout) {
    const int e = blockIdx.x;
    const int k = threadIdx.x;
    if (e < 512)       wcat[e * 128 + k]        = (h16)wlr[e * 128 + k];
    else if (e < 640)  wcat[e * 128 + k]        = (h16)wgate[(e - 512) * 128 + k];
    else               wout[(e - 640) * 128 + k] = (h16)wout_in[(e - 640) * 128 + k];
}

// ---------------------------------------------------------------------------
// k_ln1: LayerNorm over D=128 per position; one wave per row; output f16.
// ---------------------------------------------------------------------------
__global__ void __launch_bounds__(256) k_ln1(const float* __restrict__ x,
                                             const float* __restrict__ gw,
                                             const float* __restrict__ bw,
                                             h16* __restrict__ xn) {
    const int row  = blockIdx.x * 8 + (threadIdx.x >> 5);
    const int lane = threadIdx.x & 31;
    const float4 xv = *(const float4*)(x + (size_t)row * 128 + lane * 4);
    float s = xv.x + xv.y + xv.z + xv.w;
#pragma unroll
    for (int off = 16; off > 0; off >>= 1) s += __shfl_xor(s, off, 32);
    const float mu = s * 0.0078125f;
    const float d0 = xv.x - mu, d1 = xv.y - mu, d2 = xv.z - mu, d3 = xv.w - mu;
    float q = d0 * d0 + d1 * d1 + d2 * d2 + d3 * d3;
#pragma unroll
    for (int off = 16; off > 0; off >>= 1) q += __shfl_xor(q, off, 32);
    const float rs = rsqrtf(q * 0.0078125f + 1e-5f);
    const float4 g4 = *(const float4*)(gw + lane * 4);
    const float4 b4 = *(const float4*)(bw + lane * 4);
    h4 y;
    y[0] = (h16)(d0 * rs * g4.x + b4.x);
    y[1] = (h16)(d1 * rs * g4.y + b4.y);
    y[2] = (h16)(d2 * rs * g4.z + b4.z);
    y[3] = (h16)(d3 * rs * g4.w + b4.w);
    *(h4*)(xn + (size_t)row * 128 + lane * 4) = y;
}

// ---------------------------------------------------------------------------
// k_gemm1: proj = xn @ Wcat^T with fused GLU / sigmoid epilogue.
// grid.x = M/128 (block row tile, all rows share i' = m0>>9)
// grid.y: 0..7  -> paired slabs: a-cols [32y,32y+32) with g-cols (+256)
//         8..11 -> gate slabs:   cols 512+32(y-8) ... (sigmoid only)
// Epilogue LDS-transposes each wave's 16x32 tile so stores to the
// channel-major [d][k][pos] left/right buffers are 32B-contiguous per lane.
// ---------------------------------------------------------------------------
__global__ void __launch_bounds__(256) k_gemm1(const h16* __restrict__ xn,
                                               const h16* __restrict__ wcat,
                                               const unsigned char* __restrict__ maskb,
                                               h16* __restrict__ left,
                                               h16* __restrict__ right,
                                               h16* __restrict__ gate) {
    __shared__ h16 st[8][16][32];
    const int tid = threadIdx.x, lane = tid & 31, w = tid >> 5;
    const int hi = lane >> 4;
    const int m0 = blockIdx.x * 128;
    const int mrow = m0 + w * 16;
    const int yi = blockIdx.y;
    const h16* arow = xn + (size_t)(mrow + (lane & 15)) * 128;

    if (yi < 8) {
        const int ca0 = yi * 32, cg0 = ca0 + 256;
        v8f accA[2] = {}, accG[2] = {};
#pragma unroll
        for (int kk = 0; kk < 128; kk += 32) {
            const v16h a = frag_a(arow, kk, hi);
#pragma unroll
            for (int nt = 0; nt < 2; ++nt) {
                const v16h bA = frag_b(wcat + (size_t)(ca0 + nt * 16 + (lane & 15)) * 128, kk, hi);
                const v16h bG = frag_b(wcat + (size_t)(cg0 + nt * 16 + (lane & 15)) * 128, kk, hi);
                accA[nt] = wmma16(a, bA, accA[nt]);
                accG[nt] = wmma16(a, bG, accG[nt]);
            }
        }
        const int iG = m0 >> 9;              // block-uniform i'
        const int jbase = (m0 & 511) + w * 16;
        const float mi_ = maskb[iG] ? 1.0f : 0.0f;
#pragma unroll
        for (int nt = 0; nt < 2; ++nt)
#pragma unroll
            for (int r = 0; r < 8; ++r) {
                const int rr = r + hi * 8;
                float v = accA[nt][r] * sigmoidf_(accG[nt][r]);
                v *= mi_ * (maskb[jbase + rr] ? 1.0f : 0.0f);
                st[w][rr][nt * 16 + (lane & 15)] = (h16)v;
            }
        __syncthreads();
        {
            const int col = lane;            // 0..31 -> channel within slab
            const int dabs = ca0 + col;      // slab never straddles 128
            h16* basep = (dabs < 128) ? (left + (size_t)dabs * MPOS)
                                      : (right + (size_t)(dabs - 128) * MPOS);
            h16* dst = basep + iG * 512 + jbase;
            union { v8h v; h16 e[8]; } u0, u1;
#pragma unroll
            for (int r = 0; r < 8; ++r) { u0.e[r] = st[w][r][col]; u1.e[r] = st[w][r + 8][col]; }
            *(v8h*)dst = u0.v;
            *(v8h*)(dst + 8) = u1.v;
        }
    } else {
        const int gd0 = (yi - 8) * 32, c0 = 512 + gd0;
        v8f acc[2] = {};
#pragma unroll
        for (int kk = 0; kk < 128; kk += 32) {
            const v16h a = frag_a(arow, kk, hi);
#pragma unroll
            for (int nt = 0; nt < 2; ++nt) {
                const v16h b = frag_b(wcat + (size_t)(c0 + nt * 16 + (lane & 15)) * 128, kk, hi);
                acc[nt] = wmma16(a, b, acc[nt]);
            }
        }
#pragma unroll
        for (int nt = 0; nt < 2; ++nt)
#pragma unroll
            for (int r = 0; r < 8; ++r) {
                const int rr = r + hi * 8;
                st[w][rr][nt * 16 + (lane & 15)] = (h16)sigmoidf_(acc[nt][r]);
            }
        __syncthreads();
        if (lane < 16) {
            const int pos = mrow + lane;
            h16* dst = gate + (size_t)pos * 128 + gd0;
#pragma unroll
            for (int c8 = 0; c8 < 4; ++c8) {
                union { v8h v; h16 e[8]; } u;
#pragma unroll
                for (int e = 0; e < 8; ++e) u.e[e] = st[w][lane][c8 * 8 + e];
                *(v8h*)(dst + c8 * 8) = u.v;
            }
        }
    }
}

// ---------------------------------------------------------------------------
// k_gemm2: the heavy batched GEMM. For each channel d:
//   O_d[i,j] = sum_k right[d][k][i] * left[d][k][j]
// Block: 256 thr / 8 waves, tile 128x128, K staged 32 at a time through LDS
// (natural [k][128] layout, ld = 132 halves: 264B rows keep 8B alignment for
// ASYNC_B64 LDS destinations, and 16*66 words == 32 mod 64 so the two K
// lane-groups of the fragment reads hit different banks).
// Wave tile 32(i) x 64(j) -> 8 wmma per K-step.
// ---------------------------------------------------------------------------
#define LDK 132
__global__ void __launch_bounds__(256) k_gemm2(const h16* __restrict__ left,
                                               const h16* __restrict__ right,
                                               h16* __restrict__ o) {
    __shared__ h16 ldsR[32 * LDK];
    __shared__ h16 ldsL[32 * LDK];
    const int tid = threadIdx.x, lane = tid & 31, w = tid >> 5;
    const int i0 = blockIdx.x * 128, j0 = blockIdx.y * 128, d = blockIdx.z;
    const int wi = w & 3, wj = w >> 2;
    const h16* Rbase = right + (size_t)d * MPOS;
    const h16* Lbase = left + (size_t)d * MPOS;

    // staging role: half the threads stage R, half stage L; 32 halves each.
    const int slab = tid >> 7;
    const int srow = (tid & 127) >> 2;
    const int scol = (tid & 3) * 32;
    const h16* gsrc0 = (slab ? (Lbase + j0) : (Rbase + i0)) + srow * 512 + scol;
    h16* lds_dst = (slab ? ldsL : ldsR) + srow * LDK + scol;

    v8f acc[2][4] = {};
    for (int k0 = 0; k0 < 512; k0 += 32) {
        __syncthreads();
        const h16* gp = gsrc0 + (size_t)k0 * 512;
        // prefetch next K-slab while this one is consumed
        __builtin_prefetch(gsrc0 + (size_t)(k0 + 32) * 512, 0, 0);
#if USE_ASYNC_LDS
        // direct global -> LDS DMA, tracked by ASYNCcnt (no VGPR round-trip)
#pragma unroll
        for (int c = 0; c < 8; ++c) {
            __builtin_amdgcn_global_load_async_to_lds_b64(
                AS1P64(gp + c * 4), AS3P64(lds_dst + c * 4), 0, 0);
        }
        __builtin_amdgcn_s_wait_asynccnt(0);
#else
        {
            const v8h t0 = *(const v8h*)(gp);
            const v8h t1 = *(const v8h*)(gp + 8);
            const v8h t2 = *(const v8h*)(gp + 16);
            const v8h t3 = *(const v8h*)(gp + 24);
            h2* ld2 = (h2*)lds_dst;
            union { v8h v; h2 p[4]; } u;
            u.v = t0; ld2[0]  = u.p[0]; ld2[1]  = u.p[1]; ld2[2]  = u.p[2]; ld2[3]  = u.p[3];
            u.v = t1; ld2[4]  = u.p[0]; ld2[5]  = u.p[1]; ld2[6]  = u.p[2]; ld2[7]  = u.p[3];
            u.v = t2; ld2[8]  = u.p[0]; ld2[9]  = u.p[1]; ld2[10] = u.p[2]; ld2[11] = u.p[3];
            u.v = t3; ld2[12] = u.p[0]; ld2[13] = u.p[1]; ld2[14] = u.p[2]; ld2[15] = u.p[3];
        }
#endif
        __syncthreads();

        // A fragments: A[m=i][k] = ldsR[k][i] (transpose-on-read per A layout)
        const int kadd = (lane & 16) ? 16 : 0;
        v16h A2[2];
#pragma unroll
        for (int mi = 0; mi < 2; ++mi) {
            const int m = wi * 32 + mi * 16 + (lane & 15);
#pragma unroll
            for (int t = 0; t < 16; ++t) {
                const int k = t + kadd;
                A2[mi][t] = ldsR[k * LDK + m];
            }
        }
        // B fragments: B[k][n=j] = ldsL[k][j]
        const int kb = (lane & 16) ? 16 : 0;
        v16h B4[4];
#pragma unroll
        for (int nj = 0; nj < 4; ++nj) {
            const int n = wj * 64 + nj * 16 + (lane & 15);
#pragma unroll
            for (int t = 0; t < 16; ++t) B4[nj][t] = ldsL[(kb + t) * LDK + n];
        }
#pragma unroll
        for (int mi = 0; mi < 2; ++mi)
#pragma unroll
            for (int nj = 0; nj < 4; ++nj)
                acc[mi][nj] = wmma16(A2[mi], B4[nj], acc[mi][nj]);
    }

    // store O_d channel-major [d][i][j] as f16 (2B x 16-lane contiguous runs)
    h16* obase = o + (size_t)d * MPOS;
#pragma unroll
    for (int mi = 0; mi < 2; ++mi)
#pragma unroll
        for (int nj = 0; nj < 4; ++nj) {
            const int jg = j0 + wj * 64 + nj * 16 + (lane & 15);
#pragma unroll
            for (int r = 0; r < 8; ++r) {
                const int ig = i0 + wi * 32 + mi * 16 + r + ((lane >> 4) << 3);
                obase[(size_t)ig * 512 + jg] = (h16)acc[mi][nj][r];
            }
        }
}

// ---------------------------------------------------------------------------
// k_ln2gate: read channel-major o, LN over 128 channels per position,
// multiply by sigmoid gate, write position-major og (f16).
// Block covers 32 consecutive positions x all 128 channels.
// ---------------------------------------------------------------------------
__global__ void __launch_bounds__(256) k_ln2gate(const h16* __restrict__ o,
                                                 const h16* __restrict__ gate,
                                                 const float* __restrict__ og_g,
                                                 const float* __restrict__ og_b,
                                                 h16* __restrict__ og) {
    __shared__ float rS[8][32], rQ[8][32], muS[32], rsS[32];
    const int tid = threadIdx.x;
    const int j = tid & 31, g8 = tid >> 5;
    const int pos = blockIdx.x * 32 + j;

    float v[16], s = 0.f, q = 0.f;
#pragma unroll
    for (int c = 0; c < 16; ++c) {
        const float t = (float)o[(size_t)(g8 * 16 + c) * MPOS + pos];
        v[c] = t; s += t; q += t * t;
    }
    rS[g8][j] = s; rQ[g8][j] = q;
    __syncthreads();
    if (g8 == 0) {
        float ts = 0.f, tq = 0.f;
#pragma unroll
        for (int e = 0; e < 8; ++e) { ts += rS[e][j]; tq += rQ[e][j]; }
        const float mu = ts * 0.0078125f;
        muS[j] = mu;
        rsS[j] = rsqrtf(tq * 0.0078125f - mu * mu + 1e-5f);
    }
    __syncthreads();
    const float mu = muS[j], rs = rsS[j];
    const h16* gp = gate + (size_t)pos * 128 + g8 * 16;
    const v8h gv0 = *(const v8h*)gp;
    const v8h gv1 = *(const v8h*)(gp + 8);
    union { v8h v; h16 e[8]; } o0, o1;
#pragma unroll
    for (int c = 0; c < 8; ++c) {
        const int ch = g8 * 16 + c;
        o0.e[c] = (h16)(((v[c] - mu) * rs * og_g[ch] + og_b[ch]) * (float)gv0[c]);
    }
#pragma unroll
    for (int c = 0; c < 8; ++c) {
        const int ch = g8 * 16 + 8 + c;
        o1.e[c] = (h16)(((v[8 + c] - mu) * rs * og_g[ch] + og_b[ch]) * (float)gv1[c]);
    }
    h16* dst = og + (size_t)pos * 128 + g8 * 16;
    *(v8h*)dst = o0.v;
    *(v8h*)(dst + 8) = o1.v;
}

// ---------------------------------------------------------------------------
// k_gemm3: out = og @ W_out^T  (M=262144, N=128, K=128), fp32 output.
// ---------------------------------------------------------------------------
__global__ void __launch_bounds__(256) k_gemm3(const h16* __restrict__ og,
                                               const h16* __restrict__ wout,
                                               float* __restrict__ out) {
    const int tid = threadIdx.x, lane = tid & 31, w = tid >> 5;
    const int hi = lane >> 4;
    const int m0 = blockIdx.x * 128;
    const int n0 = blockIdx.y * 64;
    const h16* arow = og + (size_t)(m0 + w * 16 + (lane & 15)) * 128;
    v8f acc[4] = {};
#pragma unroll
    for (int kk = 0; kk < 128; kk += 32) {
        const v16h a = frag_a(arow, kk, hi);
#pragma unroll
        for (int nt = 0; nt < 4; ++nt) {
            const v16h b = frag_b(wout + (size_t)(n0 + nt * 16 + (lane & 15)) * 128, kk, hi);
            acc[nt] = wmma16(a, b, acc[nt]);
        }
    }
#pragma unroll
    for (int nt = 0; nt < 4; ++nt) {
        const int col = n0 + nt * 16 + (lane & 15);
#pragma unroll
        for (int r = 0; r < 8; ++r) {
            const int row = m0 + w * 16 + r + hi * 8;
            out[(size_t)row * 128 + col] = acc[nt][r];
        }
    }
}

// ---------------------------------------------------------------------------
extern "C" void kernel_launch(void* const* d_in, const int* in_sizes, int n_in,
                              void* d_out, int out_size, void* d_ws, size_t ws_size,
                              hipStream_t stream) {
    (void)in_sizes; (void)n_in; (void)out_size; (void)ws_size;
    const float*         x       = (const float*)d_in[0];
    const unsigned char* mask    = (const unsigned char*)d_in[1];
    const float*         norm_g  = (const float*)d_in[2];
    const float*         norm_b  = (const float*)d_in[3];
    const float*         W_lr    = (const float*)d_in[4];
    const float*         W_gate  = (const float*)d_in[5];
    const float*         onorm_g = (const float*)d_in[6];
    const float*         onorm_b = (const float*)d_in[7];
    const float*         W_out   = (const float*)d_in[8];
    float* out = (float*)d_out;

    char* ws = (char*)d_ws;
    const size_t SZ = (size_t)MPOS * 128 * sizeof(h16);   // 64 MiB per buffer
    h16* xn    = (h16*)(ws + 0 * SZ);   // dead after k_gemm1
    h16* left  = (h16*)(ws + 1 * SZ);
    h16* right = (h16*)(ws + 2 * SZ);
    h16* gate  = (h16*)(ws + 3 * SZ);
    h16* o     = (h16*)(ws + 4 * SZ);
    h16* wcat  = (h16*)(ws + 5 * SZ);
    h16* wout  = wcat + 640 * 128;
    h16* og    = xn;                    // alias: xn dead before og is written

    k_pack<<<768, 128, 0, stream>>>(W_lr, W_gate, W_out, wcat, wout);
    k_ln1<<<MPOS / 8, 256, 0, stream>>>(x, norm_g, norm_b, xn);
    k_gemm1<<<dim3(MPOS / 128, 12), 256, 0, stream>>>(xn, wcat, mask, left, right, gate);
    k_gemm2<<<dim3(4, 4, 128), 256, 0, stream>>>(left, right, o);
    k_ln2gate<<<MPOS / 32, 256, 0, stream>>>(o, gate, onorm_g, onorm_b, og);
    k_gemm3<<<dim3(MPOS / 128, 2), 256, 0, stream>>>(og, wout, out);
}
